// GNN_90546500534502
// MI455X (gfx1250) — matrix operand order repaired and checked by reading, not compiled, not run on hardware
//
#include <hip/hip_runtime.h>

#define N_NODES 8192
#define N_EDGES 131072
#define IN_CH   512
#define HID_CH  1024
#define NC      3
#define HEADS   8

typedef __attribute__((ext_vector_type(16))) __bf16 bf16x16;
typedef __attribute__((ext_vector_type(8)))  float  floatx8;

// ---------------- elementwise / prep kernels ----------------

__global__ void init_val_kernel(float* __restrict__ p, float v, int n) {
    int i = blockIdx.x * blockDim.x + threadIdx.x;
    if (i < n) p[i] = v;
}

__global__ void deg_kernel(const int* __restrict__ dst, float* __restrict__ deg, int e) {
    int i = blockIdx.x * blockDim.x + threadIdx.x;
    if (i < e) atomicAdd(&deg[dst[i]], 1.0f);
}

__global__ void dinv_kernel(const float* __restrict__ deg, float* __restrict__ dinv, int n) {
    int i = blockIdx.x * blockDim.x + threadIdx.x;
    if (i < n) dinv[i] = rsqrtf(deg[i]);   // deg >= 1 (self loop)
}

__global__ void cvt_bf16_kernel(const float* __restrict__ s, __bf16* __restrict__ d, int n) {
    int i = blockIdx.x * blockDim.x + threadIdx.x;
    if (i < n) d[i] = (__bf16)s[i];
}

// W [K x Nn] row-major f32  ->  Wt [Nn x K] row-major bf16
__global__ void cvt_trans_bf16_kernel(const float* __restrict__ W, __bf16* __restrict__ Wt,
                                      int K, int Nn) {
    int idx = blockIdx.x * blockDim.x + threadIdx.x;
    if (idx >= K * Nn) return;
    int n = idx / K;
    int k = idx - n * K;
    Wt[idx] = (__bf16)W[(long)k * Nn + n];
}

__global__ void init_bias_kernel(float* __restrict__ X, const float* __restrict__ b,
                                 int total, int F) {
    int idx = blockIdx.x * blockDim.x + threadIdx.x;
    if (idx < total) X[idx] = b[idx % F];
}

__global__ void self_loop_kernel(float* __restrict__ X, const float* __restrict__ H,
                                 const float* __restrict__ dinv, int total, int F) {
    int idx = blockIdx.x * blockDim.x + threadIdx.x;
    if (idx >= total) return;
    int i = idx / F;
    float dv = dinv[i];
    X[idx] += H[idx] * dv * dv;
}

__global__ void relu_kernel(float* __restrict__ X, int n) {
    int i = blockIdx.x * blockDim.x + threadIdx.x;
    if (i < n) X[i] = fmaxf(X[i], 0.0f);
}

// one workgroup per edge, streams F features, f32 global atomics
__global__ void edge_scatter_big_kernel(float* __restrict__ X, const float* __restrict__ H,
                                        const int* __restrict__ src, const int* __restrict__ dst,
                                        const float* __restrict__ dinv, int F) {
    int e = blockIdx.x;
    int s = src[e], d = dst[e];
    float w = dinv[s] * dinv[d];
    const float* hs = H + (long)s * F;
    float*       xd = X + (long)d * F;
    for (int f = threadIdx.x; f < F; f += blockDim.x)
        atomicAdd(&xd[f], hs[f] * w);
}

__global__ void edge_scatter_small_kernel(float* __restrict__ Y, const float* __restrict__ T,
                                          const int* __restrict__ src, const int* __restrict__ dst,
                                          const float* __restrict__ dinv, int e) {
    int i = blockIdx.x * blockDim.x + threadIdx.x;
    if (i >= e) return;
    int s = src[i], d = dst[i];
    float w = dinv[s] * dinv[d];
    atomicAdd(&Y[d * 3 + 0], T[s * 3 + 0] * w);
    atomicAdd(&Y[d * 3 + 1], T[s * 3 + 1] * w);
    atomicAdd(&Y[d * 3 + 2], T[s * 3 + 2] * w);
}

// ---------------- bf16 WMMA fragment loaders (wave32 layouts, ISA 7.12.2) ----------------

// A 16x32 bf16: lane m = lane&15, K-base (lane>>4)*8, elems (2j,2j+1) contiguous pairs
__device__ __forceinline__ bf16x16 frag_a(const __bf16* __restrict__ arow, int k0, int kb) {
    union { bf16x16 v; __bf16 e[16]; } a;
#pragma unroll
    for (int j = 0; j < 8; ++j) {
        int k = k0 + ((j & 4) ? 16 : 0) + kb + 2 * (j & 3);
        a.e[2 * j]     = arow[k];
        a.e[2 * j + 1] = arow[k + 1];
    }
    return a.v;
}

// B 32x16 bf16 fed from pre-transposed Bt[N x K]: lane n = lane&15, K-base (lane>>4)*8,
// elems 0..7 at K-base+i, elems 8..15 at 16+K-base+(i-8) (same order as A)
__device__ __forceinline__ bf16x16 frag_b(const __bf16* __restrict__ brow, int k0, int kb) {
    union { bf16x16 v; __bf16 e[16]; } b;
#pragma unroll
    for (int i = 0; i < 16; ++i) b.e[i] = brow[k0 + kb + ((i & 8) ? 16 : 0) + (i & 7)];
    return b.v;
}

// ---------------- bf16 WMMA GEMM, 2x2 register-blocked (32x32 output per wave) ----------
//   C[M x Nn] = A[M x K] * Bt[Nn x K]^T     (M%32==0, Nn%32==0, K%32==0)

__global__ void gemm_wmma_bf16_2x2(const __bf16* __restrict__ A, const __bf16* __restrict__ Bt,
                                   float* __restrict__ C, int M, int Nn, int K) {
    const int lane = threadIdx.x & 31;
    const int wid  = threadIdx.x >> 5;
    const int nblk = Nn >> 5;                       // 32-wide N blocks
    int w  = blockIdx.x * (blockDim.x >> 5) + wid;
    int bm = w / nblk;
    int bn = w - bm * nblk;
    if (bm * 32 >= M) return;

    const int kbA = (lane >> 4) * 8;
    const int kbB = (lane >> 4) * 8;

    const __bf16* arow0 = A  + (long)(bm * 32 + (lane & 15)) * K;
    const __bf16* arow1 = arow0 + (long)16 * K;
    const __bf16* brow0 = Bt + (long)(bn * 32 + (lane & 15)) * K;
    const __bf16* brow1 = brow0 + (long)16 * K;

    union { floatx8 v; float e[8]; } acc[2][2];
#pragma unroll
    for (int i = 0; i < 2; ++i)
#pragma unroll
        for (int j = 0; j < 2; ++j)
            acc[i][j].v = (floatx8){0.f, 0.f, 0.f, 0.f, 0.f, 0.f, 0.f, 0.f};

    for (int k0 = 0; k0 < K; k0 += 32) {
        bf16x16 a0 = frag_a(arow0, k0, kbA);
        bf16x16 a1 = frag_a(arow1, k0, kbA);
        bf16x16 b0 = frag_b(brow0, k0, kbB);
        bf16x16 b1 = frag_b(brow1, k0, kbB);
        acc[0][0].v = __builtin_amdgcn_wmma_f32_16x16x32_bf16(false, a0, false, b0, (short)0, acc[0][0].v, false, false);
        acc[0][1].v = __builtin_amdgcn_wmma_f32_16x16x32_bf16(false, a0, false, b1, (short)0, acc[0][1].v, false, false);
        acc[1][0].v = __builtin_amdgcn_wmma_f32_16x16x32_bf16(false, a1, false, b0, (short)0, acc[1][0].v, false, false);
        acc[1][1].v = __builtin_amdgcn_wmma_f32_16x16x32_bf16(false, a1, false, b1, (short)0, acc[1][1].v, false, false);
    }

    // D layout: VGPR r -> row r + (lane>=16 ? 8 : 0), col = lane&15
#pragma unroll
    for (int i = 0; i < 2; ++i) {
        const int rbase = bm * 32 + i * 16 + ((lane >> 4) ? 8 : 0);
#pragma unroll
        for (int j = 0; j < 2; ++j) {
            const int col = bn * 32 + j * 16 + (lane & 15);
#pragma unroll
            for (int r = 0; r < 8; ++r)
                C[(long)(rbase + r) * Nn + col] = acc[i][j].e[r];
        }
    }
}

// ---------------- narrow GEMM (K x 3): one wave per row ----------------

__global__ void gemm_small3_kernel(const float* __restrict__ X, const float* __restrict__ W,
                                   float* __restrict__ T, int K) {
    int lane = threadIdx.x & 31;
    int row  = blockIdx.x * (blockDim.x >> 5) + (threadIdx.x >> 5);
    const float* xr = X + (long)row * K;
    float a0 = 0.f, a1 = 0.f, a2 = 0.f;
    for (int k = lane; k < K; k += 32) {
        float xv = xr[k];
        a0 += xv * W[k * 3 + 0];
        a1 += xv * W[k * 3 + 1];
        a2 += xv * W[k * 3 + 2];
    }
    for (int off = 16; off > 0; off >>= 1) {
        a0 += __shfl_down(a0, off, 32);
        a1 += __shfl_down(a1, off, 32);
        a2 += __shfl_down(a2, off, 32);
    }
    if (lane == 0) {
        T[row * 3 + 0] = a0;
        T[row * 3 + 1] = a1;
        T[row * 3 + 2] = a2;
    }
}

// ---------------- structure-KL: one block per node ----------------

__global__ void kl_kernel(const float* __restrict__ X, const float* __restrict__ att,
                          float* __restrict__ klacc) {
    __shared__ float red[256];
    int node = blockIdx.x;
    const float* xr = X   + (long)node * HID_CH;
    const float* ar = att + (node & (HEADS - 1)) * HID_CH;
    float s = 0.f;
    for (int c = threadIdx.x; c < HID_CH; c += 256)
        s += xr[c] * ar[c];
    red[threadIdx.x] = s;
    __syncthreads();
    for (int st = 128; st > 0; st >>= 1) {
        if (threadIdx.x < st) red[threadIdx.x] += red[threadIdx.x + st];
        __syncthreads();
    }
    if (threadIdx.x == 0) {
        float a = red[0] * (1.0f / (float)HID_CH);
        a = (a >= 0.f) ? a : 0.2f * a;                      // leaky_relu(0.2)
        float p = 1.0f / (1.0f + __expf(-a));               // sigmoid
        p = fminf(fmaxf(p, 0.01f), 0.99f);                  // clip
        float term = p * __logf(2.0f * p) + (1.0f - p) * __logf(2.0f * (1.0f - p));
        atomicAdd(klacc, term);
    }
}

__global__ void write_kl_kernel(float* __restrict__ out, const float* __restrict__ klacc) {
    if (blockIdx.x == 0 && threadIdx.x == 0) out[0] = 0.5f * klacc[0];
}

// ---------------- driver ----------------

extern "C" void kernel_launch(void* const* d_in, const int* in_sizes, int n_in,
                              void* d_out, int out_size, void* d_ws, size_t ws_size,
                              hipStream_t stream) {
    const float* x    = (const float*)d_in[0];
    const int*   ei   = (const int*)d_in[1];
    const float* W1   = (const float*)d_in[2];
    const float* b1   = (const float*)d_in[3];
    const float* att1 = (const float*)d_in[4];
    const float* W2   = (const float*)d_in[5];
    const float* b2   = (const float*)d_in[6];
    const float* W3   = (const float*)d_in[7];
    const float* b3   = (const float*)d_in[8];
    const float* att2 = (const float*)d_in[9];
    const float* W4   = (const float*)d_in[10];
    const float* b4   = (const float*)d_in[11];

    const int* src = ei;
    const int* dst = ei + N_EDGES;

    float* out = (float*)d_out;
    float* y1 = out;                  // [N,3]
    float* y2 = out + 3 * N_NODES;    // [N,3]
    float* kl = out + 6 * N_NODES;    // scalar

    // workspace layout
    float* deg   = (float*)d_ws;                         // N
    float* dinv  = deg  + N_NODES;                       // N
    float* klacc = dinv + N_NODES;                       // 16 (pad)
    float* H     = klacc + 16;                           // N*HID (pre-aggregation)
    float* X1    = H  + (long)N_NODES * HID_CH;          // N*HID
    float* X2    = X1 + (long)N_NODES * HID_CH;          // N*HID
    float* T     = X2 + (long)N_NODES * HID_CH;          // N*4 (pad of N*3)
    __bf16* xb   = (__bf16*)(T + (long)N_NODES * 4);     // N*IN
    __bf16* x1b  = xb  + (long)N_NODES * IN_CH;          // N*HID
    __bf16* W1t  = x1b + (long)N_NODES * HID_CH;         // HID*IN (transposed)
    __bf16* W3t  = W1t + (long)IN_CH * HID_CH;           // HID*HID (transposed)

    const int TB = 256;
    const long NH = (long)N_NODES * HID_CH;
    dim3 blkNH((unsigned)((NH + TB - 1) / TB));
    dim3 blkN3((unsigned)((N_NODES * 3 + TB - 1) / TB));
    dim3 blkE((unsigned)((N_EDGES + TB - 1) / TB));
    // 32x32 output per wave, 8 waves per block
    const int gemm_blocks = (N_NODES / 32) * (HID_CH / 32) / 8;

    // ---- gcn_norm ----
    init_val_kernel<<<(N_NODES + TB - 1) / TB, TB, 0, stream>>>(deg, 1.0f, N_NODES);
    init_val_kernel<<<1, 32, 0, stream>>>(klacc, 0.0f, 1);
    deg_kernel<<<blkE, TB, 0, stream>>>(dst, deg, N_EDGES);
    dinv_kernel<<<(N_NODES + TB - 1) / TB, TB, 0, stream>>>(deg, dinv, N_NODES);

    // ---- weight/activation conversion ----
    cvt_bf16_kernel<<<(N_NODES * IN_CH + TB - 1) / TB, TB, 0, stream>>>(x, xb, N_NODES * IN_CH);
    cvt_trans_bf16_kernel<<<(IN_CH * HID_CH + TB - 1) / TB, TB, 0, stream>>>(W1, W1t, IN_CH, HID_CH);
    cvt_trans_bf16_kernel<<<(HID_CH * HID_CH + TB - 1) / TB, TB, 0, stream>>>(W3, W3t, HID_CH, HID_CH);

    // ---- layer 1: x1 = relu(conv(x, W1, b1)) ----
    gemm_wmma_bf16_2x2<<<gemm_blocks, TB, 0, stream>>>(xb, W1t, H, N_NODES, HID_CH, IN_CH);
    init_bias_kernel<<<blkNH, TB, 0, stream>>>(X1, b1, (int)NH, HID_CH);
    self_loop_kernel<<<blkNH, TB, 0, stream>>>(X1, H, dinv, (int)NH, HID_CH);
    edge_scatter_big_kernel<<<N_EDGES, TB, 0, stream>>>(X1, H, src, dst, dinv, HID_CH);
    relu_kernel<<<blkNH, TB, 0, stream>>>(X1, (int)NH);

    // ---- kl1 ----
    kl_kernel<<<N_NODES, 256, 0, stream>>>(X1, att1, klacc);

    // ---- y1 = conv(x1, W2, b2) ----
    gemm_small3_kernel<<<N_NODES / 8, TB, 0, stream>>>(X1, W2, T, HID_CH);
    init_bias_kernel<<<blkN3, TB, 0, stream>>>(y1, b2, N_NODES * 3, 3);
    self_loop_kernel<<<blkN3, TB, 0, stream>>>(y1, T, dinv, N_NODES * 3, 3);
    edge_scatter_small_kernel<<<blkE, TB, 0, stream>>>(y1, T, src, dst, dinv, N_EDGES);

    // ---- layer 2: x2 = relu(conv(x1, W3, b3)) ----
    cvt_bf16_kernel<<<blkNH, TB, 0, stream>>>(X1, x1b, (int)NH);
    gemm_wmma_bf16_2x2<<<gemm_blocks, TB, 0, stream>>>(x1b, W3t, H, N_NODES, HID_CH, HID_CH);
    init_bias_kernel<<<blkNH, TB, 0, stream>>>(X2, b3, (int)NH, HID_CH);
    self_loop_kernel<<<blkNH, TB, 0, stream>>>(X2, H, dinv, (int)NH, HID_CH);
    edge_scatter_big_kernel<<<N_EDGES, TB, 0, stream>>>(X2, H, src, dst, dinv, HID_CH);
    relu_kernel<<<blkNH, TB, 0, stream>>>(X2, (int)NH);

    // ---- kl2 ----
    kl_kernel<<<N_NODES, 256, 0, stream>>>(X2, att2, klacc);

    // ---- y2 = conv(x2, W4, b4) ----
    gemm_small3_kernel<<<N_NODES / 8, TB, 0, stream>>>(X2, W4, T, HID_CH);
    init_bias_kernel<<<blkN3, TB, 0, stream>>>(y2, b4, N_NODES * 3, 3);
    self_loop_kernel<<<blkN3, TB, 0, stream>>>(y2, T, dinv, N_NODES * 3, 3);
    edge_scatter_small_kernel<<<blkE, TB, 0, stream>>>(y2, T, src, dst, dinv, N_EDGES);

    // ---- scalar output ----
    write_kl_kernel<<<1, 1, 0, stream>>>(kl, klacc);
}
